// PAM_16415365005930
// MI455X (gfx1250) — matrix operand order repaired and checked by reading, not compiled, not run on hardware
//
#include <hip/hip_runtime.h>

#define Bsz 8
#define Cc  512
#define C8  64
#define HW  4096

typedef __attribute__((ext_vector_type(16))) _Float16 v16h;
typedef __attribute__((ext_vector_type(8)))  _Float16 v8h;
typedef __attribute__((ext_vector_type(8)))  float    v8f;

// ---------- WMMA helper (CDNA5 v_wmma_f32_16x16x32_f16) ----------
__device__ __forceinline__ v8f wmma_f16(v16h a, v16h b, v8f c) {
  return __builtin_amdgcn_wmma_f32_16x16x32_f16(false, a, false, b, (short)0, c, false, false);
}

__device__ __forceinline__ v8h ld8(const _Float16* p) { return *(const v8h*)p; }

__device__ __forceinline__ v16h cat8(v8h a, v8h b) {
  return __builtin_shufflevector(a, b, 0, 1, 2, 3, 4, 5, 6, 7, 8, 9, 10, 11, 12, 13, 14, 15);
}

// A-matrix 16x32 f16 from row-major [M][K]: lane m holds row m; two contiguous 16B chunks
__device__ __forceinline__ v16h load_a_tile(const _Float16* base, int ld, int row0, int k0, int lane) {
  int m = lane & 15, hi = lane >> 4;
  const _Float16* r = base + (size_t)(row0 + m) * ld + k0 + hi * 8;
  return cat8(ld8(r), ld8(r + 16));
}

// B-matrix 32x16 f16 from [N][K]-major storage (B[k][n] = src[n][k]): two contiguous 16B chunks/lane
__device__ __forceinline__ v16h load_b_tr(const _Float16* base, int ld, int k0, int col0, int lane) {
  int n = lane & 15, hi = lane >> 4;
  const _Float16* r = base + (size_t)(col0 + n) * ld + k0 + hi * 8;
  return cat8(ld8(r), ld8(r + 16));
}

// ---------- CDNA5 async global->LDS (ASYNCcnt) ----------
__device__ __forceinline__ unsigned lds_addr(const void* p) {
  return (unsigned)(size_t)p;  // low 32 bits of generic pointer = LDS byte offset
}
__device__ __forceinline__ void async_ld_b128(unsigned lds_off, const void* gaddr) {
  asm volatile("global_load_async_to_lds_b128 %0, %1, off"
               :
               : "v"(lds_off), "v"((unsigned long long)(size_t)gaddr)
               : "memory");
}
__device__ __forceinline__ void wait_async0() {
  asm volatile("s_wait_asynccnt 0" ::: "memory");
}

// ---------- Kernel 0: Wd f32 -> f16 ----------
__global__ __launch_bounds__(256) void k_convw(const float* __restrict__ Wd,
                                               _Float16* __restrict__ Wdh) {
  const int nw = Cc * Cc;
  for (int i = blockIdx.x * blockDim.x + threadIdx.x; i < nw; i += gridDim.x * blockDim.x)
    Wdh[i] = (_Float16)Wd[i];
}

// ---------- Kernel 1: tiled transpose+convert  x[b][c][n] f32 -> xt[b][n][c] f16 ----------
__global__ __launch_bounds__(256) void k_xpose(const float* __restrict__ x,
                                               _Float16* __restrict__ xt) {
  __shared__ _Float16 T[64][72];  // padded rows vs bank conflicts
  const int t = threadIdx.x;
  const int b = blockIdx.z, c0 = blockIdx.y * 64, n0 = blockIdx.x * 64;
  {
    int row = t >> 2, seg = t & 3;  // 4 threads/row, 16 floats each
    const float* src = x + ((size_t)(b * Cc + c0 + row)) * HW + n0 + seg * 16;
    v8h h0, h1;
#pragma unroll
    for (int u = 0; u < 8; ++u) h0[u] = (_Float16)src[u];
#pragma unroll
    for (int u = 0; u < 8; ++u) h1[u] = (_Float16)src[8 + u];
    *(v8h*)&T[row][seg * 16] = h0;
    *(v8h*)&T[row][seg * 16 + 8] = h1;
  }
  __syncthreads();
  {
    int nn = t >> 2, sg = t & 3;
    v8h o0, o1;
#pragma unroll
    for (int u = 0; u < 8; ++u) o0[u] = T[sg * 16 + u][nn];
#pragma unroll
    for (int u = 0; u < 8; ++u) o1[u] = T[sg * 16 + 8 + u][nn];
    _Float16* dst = xt + ((size_t)b * HW + n0 + nn) * Cc + c0 + sg * 16;
    *(v8h*)dst = o0;
    *(v8h*)(dst + 8) = o1;
  }
}

// ---------- Kernel 2: fb/fc projections -> Q[pos][64], K[pos][64] (f16, transposed) ----------
__global__ __launch_bounds__(256) void k_proj_qk(const float* __restrict__ x,
                                                 const float* __restrict__ Wb, const float* __restrict__ bb,
                                                 const float* __restrict__ Wc, const float* __restrict__ bc,
                                                 _Float16* __restrict__ Qh, _Float16* __restrict__ Kh) {
  __shared__ float xs[64][16];
  __shared__ float wbs[64][64];
  __shared__ float wcs[64][64];
  const int t = threadIdx.x;
  const int b = blockIdx.y;
  const int n0 = blockIdx.x * 16;
  const int k  = t & 63;
  const int nb = t >> 6;  // 0..3
  float accQ[4] = {0.f, 0.f, 0.f, 0.f};
  float accK[4] = {0.f, 0.f, 0.f, 0.f};
  for (int c0 = 0; c0 < Cc; c0 += 64) {
    {
      int row = t >> 2, seg = t & 3;
      ((uint4*)&xs[row][0])[seg] =
          ((const uint4*)(x + ((size_t)(b * Cc + c0 + row)) * HW + n0))[seg];
#pragma unroll
      for (int u = 0; u < 4; ++u) {
        int idx = t + u * 256;
        int rk = idx >> 4, sc = idx & 15;
        ((uint4*)&wbs[rk][0])[sc] = ((const uint4*)(Wb + (size_t)rk * Cc + c0))[sc];
        ((uint4*)&wcs[rk][0])[sc] = ((const uint4*)(Wc + (size_t)rk * Cc + c0))[sc];
      }
    }
    __syncthreads();
#pragma unroll 4
    for (int cc = 0; cc < 64; ++cc) {
      float wq = wbs[k][cc], wk = wcs[k][cc];
#pragma unroll
      for (int u = 0; u < 4; ++u) {
        float xv = xs[cc][nb + u * 4];
        accQ[u] += wq * xv;
        accK[u] += wk * xv;
      }
    }
    __syncthreads();
  }
  float bq = bb[k], bk = bc[k];
#pragma unroll
  for (int u = 0; u < 4; ++u) {
    int n = n0 + nb + u * 4;
    Qh[((size_t)b * HW + n) * C8 + k] = (_Float16)(accQ[u] + bq);
    Kh[((size_t)b * HW + n) * C8 + k] = (_Float16)(accK[u] + bk);
  }
}

// ---------- Kernel 3: fd projection via WMMA, async double-buffered staging ----------
__global__ __launch_bounds__(256) void k_proj_v(const _Float16* __restrict__ xt,
                                                const _Float16* __restrict__ Wdh,
                                                const float* __restrict__ bd,
                                                _Float16* __restrict__ Vh) {
  __shared__ _Float16 Ws[2][64 * 32];  // [c][ci]  (A source)
  __shared__ _Float16 Xs[2][64 * 32];  // [n][ci]  (B source, [N][K]-major)
  const int t = threadIdx.x, lane = t & 31, w = t >> 5;
  const int b = blockIdx.z, c0 = blockIdx.y * 64, n0 = blockIdx.x * 64;
  const int mt = (w & 3) * 16, nt = (w >> 2) * 32;
  const int row = t >> 2, seg = t & 3;
  v8f acc0 = {}, acc1 = {};

  const _Float16* wsrc = Wdh + (size_t)(c0 + row) * Cc + seg * 8;
  const _Float16* xsrc = xt + ((size_t)(b * HW + n0 + row)) * Cc + seg * 8;
  const unsigned dW0 = lds_addr(&Ws[0][row * 32 + seg * 8]);
  const unsigned dW1 = lds_addr(&Ws[1][row * 32 + seg * 8]);
  const unsigned dX0 = lds_addr(&Xs[0][row * 32 + seg * 8]);
  const unsigned dX1 = lds_addr(&Xs[1][row * 32 + seg * 8]);

  async_ld_b128(dW0, wsrc);
  async_ld_b128(dX0, xsrc);
  wait_async0();
  __syncthreads();
  for (int ci = 0; ci < Cc; ci += 32) {
    int cur = (ci >> 5) & 1;
    if (ci + 32 < Cc) {  // kick next chunk while we compute
      async_ld_b128(cur ? dW0 : dW1, wsrc + ci + 32);
      async_ld_b128(cur ? dX0 : dX1, xsrc + ci + 32);
    }
    v16h A  = load_a_tile(Ws[cur], 32, mt, 0, lane);
    v16h B0 = load_b_tr(Xs[cur], 32, 0, nt, lane);
    v16h B1 = load_b_tr(Xs[cur], 32, 0, nt + 16, lane);
    acc0 = wmma_f16(A, B0, acc0);
    acc1 = wmma_f16(A, B1, acc1);
    wait_async0();
    __syncthreads();
  }
  const int hi = lane >> 4, n = lane & 15;
#pragma unroll
  for (int r = 0; r < 8; ++r) {
    int c = c0 + mt + r + hi * 8;
    float bias = bd[c];
    _Float16* vp = Vh + ((size_t)(b * Cc + c)) * HW + n0 + nt + n;
    vp[0]  = (_Float16)(acc0[r] + bias);
    vp[16] = (_Float16)(acc1[r] + bias);
  }
}

// ---------- Kernel 4: softmax row stats via WMMA, async double-buffered K staging ----------
__global__ __launch_bounds__(256) void k_stats(const _Float16* __restrict__ Qh,
                                               const _Float16* __restrict__ Kh,
                                               float* __restrict__ mrow, float* __restrict__ lrow) {
  __shared__ _Float16 Qs[32 * 64];
  __shared__ _Float16 Ks[2][128 * 64];
  __shared__ float Zb[32 * 128];
  __shared__ float pm[32 * 8], ps[32 * 8];
  const int t = threadIdx.x, lane = t & 31, w = t >> 5;
  const int b = blockIdx.y, i0 = blockIdx.x * 32;

  const _Float16* ksrc = Kh + (size_t)b * HW * C8;
  // prologue: async-stage Q strip and first K chunk
  async_ld_b128(lds_addr(&Qs[t * 8]), Qh + ((size_t)b * HW + i0) * C8 + t * 8);
#pragma unroll
  for (int u = 0; u < 4; ++u)
    async_ld_b128(lds_addr(&Ks[0][(t + u * 256) * 8]), ksrc + (t + u * 256) * 8);
  wait_async0();
  __syncthreads();

  float m_run = -3.0e38f, l_run = 0.f;
  for (int j0 = 0; j0 < HW; j0 += 128) {
    int cur = (j0 >> 7) & 1;
    if (j0 + 128 < HW) {
#pragma unroll
      for (int u = 0; u < 4; ++u)
        async_ld_b128(lds_addr(&Ks[cur ^ 1][(t + u * 256) * 8]),
                      ksrc + (size_t)(j0 + 128) * C8 + (t + u * 256) * 8);
    }
#pragma unroll
    for (int rt = 0; rt < 2; ++rt) {
      v8f z = {};
      v16h A0 = load_a_tile(Qs, 64, rt * 16, 0, lane);
      v16h A1 = load_a_tile(Qs, 64, rt * 16, 32, lane);
      v16h B0 = load_b_tr(Ks[cur], 64, 0, w * 16, lane);
      v16h B1 = load_b_tr(Ks[cur], 64, 32, w * 16, lane);
      z = wmma_f16(A0, B0, z);
      z = wmma_f16(A1, B1, z);
      int hi = lane >> 4, n = lane & 15;
#pragma unroll
      for (int r = 0; r < 8; ++r)
        Zb[(rt * 16 + r + hi * 8) * 128 + w * 16 + n] = z[r];
    }
    __syncthreads();
    {  // partial row reductions: 8 groups of 16 columns, vectorized reads
      int r = t & 31, g = t >> 5;
      const float4* row4 = (const float4*)&Zb[r * 128 + g * 16];
      float vals[16];
#pragma unroll
      for (int u = 0; u < 4; ++u) {
        float4 q = row4[u];
        vals[u * 4 + 0] = q.x; vals[u * 4 + 1] = q.y;
        vals[u * 4 + 2] = q.z; vals[u * 4 + 3] = q.w;
      }
      float lm = -3.0e38f;
#pragma unroll
      for (int u = 0; u < 16; ++u) lm = fmaxf(lm, vals[u]);
      float ls = 0.f;
#pragma unroll
      for (int u = 0; u < 16; ++u) ls += __expf(vals[u] - lm);
      pm[r * 8 + g] = lm;
      ps[r * 8 + g] = ls;
    }
    __syncthreads();
    if (t < 32) {  // online combine across chunk
      float cm = pm[t * 8];
#pragma unroll
      for (int g = 1; g < 8; ++g) cm = fmaxf(cm, pm[t * 8 + g]);
      float cs = 0.f;
#pragma unroll
      for (int g = 0; g < 8; ++g) cs += ps[t * 8 + g] * __expf(pm[t * 8 + g] - cm);
      float nm = fmaxf(m_run, cm);
      l_run = l_run * __expf(m_run - nm) + cs * __expf(cm - nm);
      m_run = nm;
    }
    wait_async0();
    __syncthreads();
  }
  if (t < 32) {
    mrow[(size_t)b * HW + i0 + t] = m_run;
    lrow[(size_t)b * HW + i0 + t] = l_run;
  }
}

// ---------- Kernel 5: fused P recompute + e = V*P + epilogue, async Q staging ----------
__global__ __launch_bounds__(256) void k_attn(const _Float16* __restrict__ Qh,
                                              const _Float16* __restrict__ Kh,
                                              const _Float16* __restrict__ Vh,
                                              const float* __restrict__ mrow,
                                              const float* __restrict__ lrow,
                                              const float* __restrict__ x,
                                              const float* __restrict__ alpha,
                                              float* __restrict__ out) {
  __shared__ _Float16 Qs[2][32 * 64];
  __shared__ _Float16 Ks[32 * 64];
  __shared__ _Float16 Pt[32 * 32];  // transposed: Pt[j][i]
  __shared__ float msh[32], lsh[32];
  const int t = threadIdx.x, lane = t & 31, w = t >> 5;
  const int b = blockIdx.y, j0 = blockIdx.x * 32;
  const int c0w = w * 64;
  v8f acc[4][2] = {};

  const _Float16* qsrc = Qh + (size_t)b * HW * C8 + t * 8;
  // prologue: async-stage K strip (used all iterations) and first Q strip
  async_ld_b128(lds_addr(&Ks[t * 8]), Kh + ((size_t)b * HW + j0) * C8 + t * 8);
  async_ld_b128(lds_addr(&Qs[0][t * 8]), qsrc);
  wait_async0();

  for (int i0 = 0; i0 < HW; i0 += 32) {
    int cur = (i0 >> 5) & 1;
    if (t < 32) {
      msh[t] = mrow[(size_t)b * HW + i0 + t];
      lsh[t] = lrow[(size_t)b * HW + i0 + t];
    }
    if (i0 + 32 < HW) {
      async_ld_b128(lds_addr(&Qs[cur ^ 1][t * 8]), qsrc + (size_t)(i0 + 32) * C8);
      __builtin_prefetch(Vh + ((size_t)(b * Cc + c0w + (lane & 3) * 16 + (lane >> 2))) * HW + i0 + 32, 0, 1);
    }
    __syncthreads();
    if (w < 4) {  // phase A: Z tile (32i x 32j) -> Pt (transposed) in LDS
      int it = (w & 1) * 16, jt = (w >> 1) * 16;
      v8f z = {};
      v16h A0 = load_a_tile(Qs[cur], 64, it, 0, lane);
      v16h A1 = load_a_tile(Qs[cur], 64, it, 32, lane);
      v16h B0 = load_b_tr(Ks, 64, 0, jt, lane);
      v16h B1 = load_b_tr(Ks, 64, 32, jt, lane);
      z = wmma_f16(A0, B0, z);
      z = wmma_f16(A1, B1, z);
      int hi = lane >> 4, n = lane & 15;
      int ri0 = it + hi * 8;  // this lane's 8 output rows are contiguous in i
      v8h ph;
#pragma unroll
      for (int r = 0; r < 8; ++r)
        ph[r] = (_Float16)(__expf(z[r] - msh[ri0 + r]) / lsh[ri0 + r]);
      *(v8h*)(Pt + (jt + n) * 32 + ri0) = ph;  // single b128 store
    }
    __syncthreads();
    // phase B: every wave accumulates its 64c x 32j slab of e
    v16h Bp0 = load_b_tr(Pt, 32, 0, 0, lane);
    v16h Bp1 = load_b_tr(Pt, 32, 0, 16, lane);
#pragma unroll
    for (int at = 0; at < 4; ++at) {
      const _Float16* vb = Vh + ((size_t)(b * Cc + c0w + at * 16)) * HW + i0;
      v16h A = load_a_tile(vb, HW, 0, 0, lane);
      acc[at][0] = wmma_f16(A, Bp0, acc[at][0]);
      acc[at][1] = wmma_f16(A, Bp1, acc[at][1]);
    }
    wait_async0();
    __syncthreads();
  }
  const float al = alpha[0];
  const int hi = lane >> 4, n = lane & 15;
#pragma unroll
  for (int at = 0; at < 4; ++at)
#pragma unroll
    for (int bt = 0; bt < 2; ++bt)
#pragma unroll
      for (int r = 0; r < 8; ++r) {
        int c = c0w + at * 16 + r + hi * 8;
        int col = j0 + bt * 16 + n;
        size_t idx = ((size_t)(b * Cc + c)) * HW + col;
        out[idx] = al * acc[at][bt][r] + x[idx];
      }
}

// ---------- host launcher ----------
extern "C" void kernel_launch(void* const* d_in, const int* in_sizes, int n_in,
                              void* d_out, int out_size, void* d_ws, size_t ws_size,
                              hipStream_t stream) {
  (void)in_sizes; (void)n_in; (void)out_size; (void)ws_size;
  const float* x     = (const float*)d_in[0];
  const float* Wb    = (const float*)d_in[1];
  const float* bb    = (const float*)d_in[2];
  const float* Wc    = (const float*)d_in[3];
  const float* bc    = (const float*)d_in[4];
  const float* Wd    = (const float*)d_in[5];
  const float* bd    = (const float*)d_in[6];
  const float* alpha = (const float*)d_in[7];
  float* out = (float*)d_out;

  char* ws = (char*)d_ws;
  size_t off = 0;
  auto carve = [&](size_t bytes) -> char* {
    char* p = ws + off;
    off += (bytes + 255) & ~(size_t)255;
    return p;
  };
  _Float16* xt  = (_Float16*)carve((size_t)Bsz * HW * Cc * 2);  // 32 MB  x^T in f16
  _Float16* Wdh = (_Float16*)carve((size_t)Cc * Cc * 2);        // 0.5 MB
  _Float16* Qh  = (_Float16*)carve((size_t)Bsz * HW * C8 * 2);  // 4 MB
  _Float16* Kh  = (_Float16*)carve((size_t)Bsz * HW * C8 * 2);  // 4 MB
  _Float16* Vh  = (_Float16*)carve((size_t)Bsz * Cc * HW * 2);  // 32 MB
  float* mrow   = (float*)carve((size_t)Bsz * HW * 4);
  float* lrow   = (float*)carve((size_t)Bsz * HW * 4);

  k_convw<<<256, 256, 0, stream>>>(Wd, Wdh);
  k_xpose<<<dim3(HW / 64, Cc / 64, Bsz), 256, 0, stream>>>(x, xt);
  k_proj_qk<<<dim3(HW / 16, Bsz), 256, 0, stream>>>(x, Wb, bb, Wc, bc, Qh, Kh);
  k_proj_v<<<dim3(HW / 64, Cc / 64, Bsz), 256, 0, stream>>>(xt, Wdh, bd, Vh);
  k_stats<<<dim3(HW / 32, Bsz), 256, 0, stream>>>(Qh, Kh, mrow, lrow);
  k_attn<<<dim3(HW / 32, Bsz), 256, 0, stream>>>(Qh, Kh, Vh, mrow, lrow, x, alpha, out);
}
